// SparseMoE_87711822119469
// MI455X (gfx1250) — matrix operand order repaired and checked by reading, not compiled, not run on hardware
//
#include <hip/hip_runtime.h>
#include <hip/hip_bf16.h>

typedef __bf16 bf16_t;
typedef __attribute__((ext_vector_type(16))) __bf16 v16bf;
typedef __attribute__((ext_vector_type(8)))  float  v8f;

#define B_TOK 32768
#define D_IN  1024
#define D_OUT 1024
#define NE    8

// GEMM tiling
#define MT   128          // block tile M (tokens)
#define NT   128          // block tile N (out features)
#define KT   32           // K step (one WMMA K)
#define NKT  (D_IN / KT)  // K iterations
#define ASTR 40           // padded LDS row stride for A (elements)
#define BSTR 136          // padded LDS row stride for B (elements)

// ---------------------------------------------------------------------------
// Load a 16-element bf16 fragment from LDS as two 16B vector loads.
// ---------------------------------------------------------------------------
__device__ __forceinline__ v16bf ld_frag(const bf16_t* p0, const bf16_t* p1) {
    union { uint4 u[2]; v16bf v; } t;
    t.u[0] = *(const uint4*)p0;
    t.u[1] = *(const uint4*)p1;
    return t.v;
}

// ---------------------------------------------------------------------------
// We (fp32) -> bf16 conversion
// ---------------------------------------------------------------------------
__global__ void moe_cvt_bf16(const float* __restrict__ in, bf16_t* __restrict__ out, int n) {
    int i = blockIdx.x * blockDim.x + threadIdx.x;
    if (i < n) out[i] = (bf16_t)in[i];
}

// ---------------------------------------------------------------------------
// Router: one wave32 per token. logits = x@Wr + br, top-2, renormalize.
// Fuses the x -> bf16 conversion (x fp32 is read exactly once).
// ---------------------------------------------------------------------------
__global__ void moe_router(const float* __restrict__ x, const float* __restrict__ Wr,
                           const float* __restrict__ br, bf16_t* __restrict__ xq,
                           float* __restrict__ gate_w, int* __restrict__ gate_i,
                           int* __restrict__ counts) {
    const int tok  = (blockIdx.x * blockDim.x + threadIdx.x) >> 5;
    const int lane = threadIdx.x & 31;
    if (tok >= B_TOK) return;
    const float* xr = x + (size_t)tok * D_IN;
    bf16_t*      xo = xq + (size_t)tok * D_IN;

    float acc[NE];
#pragma unroll
    for (int e = 0; e < NE; ++e) acc[e] = 0.0f;

#pragma unroll 4
    for (int j = 0; j < D_IN / 32; ++j) {
        int d = lane + 32 * j;
        float xv = xr[d];
        xo[d] = (bf16_t)xv;
        const float* wr = Wr + d * NE;
#pragma unroll
        for (int e = 0; e < NE; ++e) acc[e] += xv * wr[e];
    }
#pragma unroll
    for (int off = 16; off > 0; off >>= 1) {
#pragma unroll
        for (int e = 0; e < NE; ++e) acc[e] += __shfl_xor(acc[e], off, 32);
    }
    if (lane == 0) {
        float l[NE];
#pragma unroll
        for (int e = 0; e < NE; ++e) l[e] = acc[e] + br[e];
        int i0 = 0;
#pragma unroll
        for (int e = 1; e < NE; ++e) if (l[e] > l[i0]) i0 = e;   // ties -> lowest idx
        int i1 = (i0 == 0) ? 1 : 0;
#pragma unroll
        for (int e = 0; e < NE; ++e) if (e != i0 && l[e] > l[i1]) i1 = e;
        // renormalized top-2 softmax == sigmoid of logit gap
        float w0 = 1.0f / (1.0f + __expf(l[i1] - l[i0]));
        float w1 = 1.0f - w0;
        gate_w[2 * tok]     = w0;
        gate_w[2 * tok + 1] = w1;
        gate_i[2 * tok]     = i0;
        gate_i[2 * tok + 1] = i1;
        atomicAdd(&counts[0 * NE + i0], 1);
        atomicAdd(&counts[1 * NE + i1], 1);
    }
}

// ---------------------------------------------------------------------------
// Exclusive scan of (slot, expert) counts -> offsets; seed cursors.
// ---------------------------------------------------------------------------
__global__ void moe_scan(const int* __restrict__ counts, int* __restrict__ offsets,
                         int* __restrict__ cursors) {
    if (threadIdx.x == 0 && blockIdx.x == 0) {
        for (int s = 0; s < 2; ++s) {
            int run = 0;
            for (int e = 0; e < NE; ++e) {
                offsets[s * NE + e] = run;
                cursors[s * NE + e] = run;
                run += counts[s * NE + e];
            }
        }
    }
}

// ---------------------------------------------------------------------------
// Build per-(slot, expert) token lists.
// ---------------------------------------------------------------------------
__global__ void moe_fill(const int* __restrict__ gate_i, int* __restrict__ cursors,
                         int* __restrict__ list) {
    int b = blockIdx.x * blockDim.x + threadIdx.x;
    if (b >= B_TOK) return;
    int e0 = gate_i[2 * b];
    int e1 = gate_i[2 * b + 1];
    int p0 = atomicAdd(&cursors[e0], 1);
    list[p0] = b;
    int p1 = atomicAdd(&cursors[NE + e1], 1);
    list[B_TOK + p1] = b;
}

// ---------------------------------------------------------------------------
// Grouped bf16 WMMA GEMM over gathered token rows of one routing slot.
// Block = 256 threads (8 waves), tile 128(M) x 128(N); wave computes 32x64.
// Double-buffered LDS: one barrier per K step; global loads of tile k+1
// overlap the 8 WMMAs of tile k.
// Fragment layouts per CDNA5 ISA 7.12.2:
//   A (16x32 bf16): lane = M row; VGPR pairs hold K in {0..7,16..23} (half 0)
//                   or {8..15,24..31} (half 1)            -> 2x ds_load_b128
//   B (32x16 bf16): lane = N col; K order same as A (table order)
//                   gathered from K-major LDS             -> 16x ds_load_u16
//   D: VGPR r -> M = r + 8*(lane>>4), N = lane&15
// slot==0: out = w*(x@We + be);  slot==1: out += w*(x@We + be)
// ---------------------------------------------------------------------------
__global__ void __launch_bounds__(256)
moe_gemm(const bf16_t* __restrict__ xq, const bf16_t* __restrict__ Weq,
         const float* __restrict__ be, const float* __restrict__ gate_w,
         const int* __restrict__ list, const int* __restrict__ counts,
         const int* __restrict__ offsets, float* __restrict__ out, int slot) {
    const int e   = blockIdx.z;
    const int cnt = counts[slot * NE + e];
    const int m0  = blockIdx.y * MT;
    if (m0 >= cnt) return;
    const int base = offsets[slot * NE + e];
    const int n0   = blockIdx.x * NT;

    __shared__ bf16_t sA[2][MT * ASTR];  // double-buffered gathered A tile
    __shared__ bf16_t sB[2][KT * BSTR];  // double-buffered B tile (K-major)
    __shared__ int    sTok[MT];
    __shared__ float  sW[MT];

    const int t    = threadIdx.x;
    const int lane = t & 31;
    const int wv   = t >> 5;             // 0..7
    const int wm   = (wv & 3) * 32;      // wave M offset in tile
    const int wn   = (wv >> 2) * 64;     // wave N offset in tile
    const int half = lane >> 4;
    const int sub  = lane & 15;

    if (t < MT) {
        int row = m0 + t;
        int cr  = row < cnt ? row : (cnt - 1);     // clamp; stores are masked
        int tok = list[slot * B_TOK + base + cr];
        sTok[t] = tok;
        sW[t]   = gate_w[2 * tok + slot];
    }
    __syncthreads();

    v8f acc[2][4];
#pragma unroll
    for (int i = 0; i < 2; ++i)
#pragma unroll
        for (int j = 0; j < 4; ++j) acc[i][j] = (v8f)(0.0f);

    const bf16_t* Be = Weq + (size_t)e * D_IN * D_OUT + n0;

    // staging coordinates (constant across K loop)
    const int ra = t >> 1,  ca = (t & 1) * 16;   // A: 128 rows x 32 cols
    const int rb = t >> 3,  cb = (t & 7) * 16;   // B:  32 rows x 128 cols

    // hoisted per-thread global source bases (loop-invariant)
    const bf16_t* aRow = xq + (size_t)sTok[ra] * D_IN + ca;   // + k0
    const bf16_t* bRow = Be + (size_t)rb * D_OUT + cb;        // + k0*D_OUT

    // stage one K tile into LDS buffer `buf`
    auto stage = [&](int buf, int k0) {
        const uint4* sa = (const uint4*)(aRow + k0);
        uint4* da = (uint4*)(&sA[buf][ra * ASTR + ca]);
        da[0] = sa[0];
        da[1] = sa[1];
        const uint4* sb = (const uint4*)(bRow + (size_t)k0 * D_OUT);
        uint4* db = (uint4*)(&sB[buf][rb * BSTR + cb]);
        db[0] = sb[0];
        db[1] = sb[1];
    };

    stage(0, 0);
    __syncthreads();

    for (int kt = 0; kt < NKT; ++kt) {
        const int cur = kt & 1;
        if (kt + 1 < NKT) {
            stage(cur ^ 1, (kt + 1) * KT);           // overlap with WMMAs below
            if (kt + 2 < NKT) {
                __builtin_prefetch(aRow + (kt + 2) * KT, 0, 1);
                __builtin_prefetch(bRow + (size_t)(kt + 2) * KT * D_OUT, 0, 1);
            }
        }

        // ---- fragments from current buffer: A 2x ds_load_b128, B 16x ds_load_u16 ----
        v16bf af[2];
#pragma unroll
        for (int i = 0; i < 2; ++i) {
            const bf16_t* pa = &sA[cur][(wm + i * 16 + sub) * ASTR + 8 * half];
            af[i] = ld_frag(pa, pa + 16);
        }
        v16bf bfr[4];
#pragma unroll
        for (int j = 0; j < 4; ++j) {
            const bf16_t* pb = &sB[cur][wn + j * 16 + sub];
#pragma unroll
            for (int s = 0; s < 16; ++s) {
                int k = (s < 8) ? (8 * half + s) : (16 + 8 * half + (s - 8));
                bfr[j][s] = pb[k * BSTR];
            }
        }
#pragma unroll
        for (int i = 0; i < 2; ++i)
#pragma unroll
            for (int j = 0; j < 4; ++j)
                acc[i][j] = __builtin_amdgcn_wmma_f32_16x16x32_bf16(
                    false, af[i], false, bfr[j], (short)0, acc[i][j], false, false);

        __syncthreads();   // next buffer complete + current buffer free
    }

    // ---- epilogue: D layout -> VGPR r: M = r + 8*half, N = sub ----
#pragma unroll
    for (int i = 0; i < 2; ++i) {
#pragma unroll
        for (int j = 0; j < 4; ++j) {
#pragma unroll
            for (int r = 0; r < 8; ++r) {
                int ml = wm + i * 16 + r + 8 * half;
                if (m0 + ml < cnt) {
                    int   tok = sTok[ml];
                    float wgt = sW[ml];
                    int   ng  = n0 + wn + j * 16 + sub;
                    float v   = wgt * (acc[i][j][r] + be[e * D_OUT + ng]);
                    size_t oi = (size_t)tok * D_OUT + ng;
                    if (slot == 0) out[oi] = v;
                    else           out[oi] += v;
                }
            }
        }
    }
}

// ---------------------------------------------------------------------------
extern "C" void kernel_launch(void* const* d_in, const int* in_sizes, int n_in,
                              void* d_out, int out_size, void* d_ws, size_t ws_size,
                              hipStream_t stream) {
    const float* x  = (const float*)d_in[0];
    const float* Wr = (const float*)d_in[1];
    const float* br = (const float*)d_in[2];
    const float* We = (const float*)d_in[3];
    const float* be = (const float*)d_in[4];
    float* out = (float*)d_out;

    char* ws = (char*)d_ws;
    bf16_t* xq      = (bf16_t*)ws;  ws += (size_t)B_TOK * D_IN * sizeof(bf16_t);
    bf16_t* Weq     = (bf16_t*)ws;  ws += (size_t)NE * D_IN * D_OUT * sizeof(bf16_t);
    float*  gw      = (float*)ws;   ws += (size_t)B_TOK * 2 * sizeof(float);
    int*    gi      = (int*)ws;     ws += (size_t)B_TOK * 2 * sizeof(int);
    int*    counts  = (int*)ws;     ws += 16 * sizeof(int);
    int*    offsets = (int*)ws;     ws += 16 * sizeof(int);
    int*    cursors = (int*)ws;     ws += 16 * sizeof(int);
    int*    list    = (int*)ws;     // 2*B_TOK ints

    hipMemsetAsync(counts, 0, 16 * sizeof(int), stream);

    moe_cvt_bf16<<<(NE * D_IN * D_OUT) / 256, 256, 0, stream>>>(We, Weq, NE * D_IN * D_OUT);
    moe_router<<<(B_TOK * 32) / 256, 256, 0, stream>>>(x, Wr, br, xq, gw, gi, counts);
    moe_scan<<<1, 1, 0, stream>>>(counts, offsets, cursors);
    moe_fill<<<B_TOK / 256, 256, 0, stream>>>(gi, cursors, list);

    dim3 grid(D_OUT / NT, B_TOK / MT, NE);
    moe_gemm<<<grid, 256, 0, stream>>>(xq, Weq, be, gw, list, counts, offsets, out, 0);
    moe_gemm<<<grid, 256, 0, stream>>>(xq, Weq, be, gw, list, counts, offsets, out, 1);
}
